// Mlp_2_Layer_73048803770370
// MI455X (gfx1250) — matrix operand order repaired and checked by reading, not compiled, not run on hardware
//
#include <hip/hip_runtime.h>

// Problem constants (from reference)
#define B_   8192
#define F_   16
#define E_   32
#define V_   100000
#define D_   8
#define H1_  1024
#define H2_  512
#define IN_  512   // F_*E_
#define CH_  16    // stats reduction chunks (B_/CH_ rows per chunk)

typedef __attribute__((ext_vector_type(16))) __bf16        v16bf;
typedef __attribute__((ext_vector_type(8)))  float          v8f;
typedef __attribute__((ext_vector_type(8)))  unsigned short v8u16;
typedef __attribute__((ext_vector_type(4)))  float          v4f;

union FragU { v16bf bf; v8u16 h[2]; };

__device__ __forceinline__ unsigned short f2bf(float f) {
  unsigned int u = __float_as_uint(f);
  // round-to-nearest-even bf16
  return (unsigned short)((u + 0x7FFFu + ((u >> 16) & 1u)) >> 16);
}
__device__ __forceinline__ float bf2f(unsigned short h) {
  return __uint_as_float(((unsigned int)h) << 16);
}

// ---------------------------------------------------------------------------
// 1) f32 -> bf16 bulk cast (weights). 8 elements per thread, vectorized.
// ---------------------------------------------------------------------------
__global__ void cast_bf16_kernel(const float* __restrict__ src,
                                 unsigned short* __restrict__ dst, size_t n8) {
  size_t i = (size_t)blockIdx.x * blockDim.x + threadIdx.x;
  if (i >= n8) return;
  const float* s = src + i * 8;
  v4f x0 = *(const v4f*)s;
  v4f x1 = *(const v4f*)(s + 4);
  v8u16 o;
  o[0]=f2bf(x0[0]); o[1]=f2bf(x0[1]); o[2]=f2bf(x0[2]); o[3]=f2bf(x0[3]);
  o[4]=f2bf(x1[0]); o[5]=f2bf(x1[1]); o[6]=f2bf(x1[2]); o[7]=f2bf(x1[3]);
  *(v8u16*)(dst + i * 8) = o;
}

// ---------------------------------------------------------------------------
// 2) Embedding gather + cast: one thread per (b, f) copies 32 f32 -> 32 bf16.
// ---------------------------------------------------------------------------
__global__ void gather_cast_kernel(const int* __restrict__ feat_ids,
                                   const float* __restrict__ tables,
                                   unsigned short* __restrict__ embB) {
  int t = blockIdx.x * blockDim.x + threadIdx.x;
  if (t >= B_ * F_) return;
  int b = t >> 4, f = t & 15;
  int id = feat_ids[t];
  const float* src = tables + ((size_t)f * V_ + (size_t)id) * E_;
  unsigned short* dst = embB + (size_t)b * IN_ + f * E_;
#pragma unroll
  for (int c = 0; c < 4; ++c) {
    v4f x0 = *(const v4f*)(src + c * 8);
    v4f x1 = *(const v4f*)(src + c * 8 + 4);
    v8u16 o;
    o[0]=f2bf(x0[0]); o[1]=f2bf(x0[1]); o[2]=f2bf(x0[2]); o[3]=f2bf(x0[3]);
    o[4]=f2bf(x1[0]); o[5]=f2bf(x1[1]); o[6]=f2bf(x1[2]); o[7]=f2bf(x1[3]);
    *(v8u16*)(dst + c * 8) = o;
  }
}

// ---------------------------------------------------------------------------
// 3) Batched bf16 GEMM: Out[d] = A[d] (MxK, row-major) * Bw[d]^T (Bw is NxK
//    row-major, i.e. already in WMMA B-operand friendly [N][K] layout).
//    One wave computes a 64x64 tile: 4x4 grid of v_wmma_f32_16x16x32_bf16.
//    Block = 256 threads = 8 waves covering 64(M) x 512(N).
//    Per-lane fragment layouts follow CDNA5 ISA 7.12.2 exactly:
//      A 16x32 bf16: lane L, M=L&15, VGPR0-3 = K kb..kb+7, VGPR4-7 = K 16+kb..
//                    (kb = (L>>4)*8)  -> two contiguous b128 loads per tile
//      B 32x16 bf16: lane L, N=L&15, VGPR0-3 = K kb..kb+7, VGPR4-7 = K 16+kb.. (as A)
//                    (kb = (L>>4)*8)  -> two contiguous b128 loads per tile
// ---------------------------------------------------------------------------
__global__ __launch_bounds__(256)
void gemm_bf16(const unsigned short* __restrict__ A, size_t aDstride,
               const unsigned short* __restrict__ Bw, size_t bDstride,
               unsigned short* __restrict__ Out, int M, int N, int K) {
  const int lane = threadIdx.x & 31;
  const int wave = threadIdx.x >> 5;
  const int d  = blockIdx.z;
  const int m0 = blockIdx.x * 64;
  const int n0 = blockIdx.y * 512 + wave * 64;
  const unsigned short* Ad = A + (size_t)d * aDstride;
  const unsigned short* Bd = Bw + (size_t)d * bDstride;

  const int lm  = lane & 15;
  const int abk = (lane >> 4) * 8;   // A K sub-offset
  const int bbk = (lane >> 4) * 8;   // B K sub-offset

  v8f acc[4][4] = {};

  for (int k = 0; k < K; k += 32) {
    FragU afr[4], bfr[4];
#pragma unroll
    for (int i = 0; i < 4; ++i) {
      const unsigned short* p = Ad + (size_t)(m0 + 16 * i + lm) * K + k + abk;
      afr[i].h[0] = *(const v8u16*)p;         // K = k+abk   .. +7
      afr[i].h[1] = *(const v8u16*)(p + 16);  // K = k+16+abk.. +7
    }
#pragma unroll
    for (int j = 0; j < 4; ++j) {
      const unsigned short* p = Bd + (size_t)(n0 + 16 * j + lm) * K + k + bbk;
      bfr[j].h[0] = *(const v8u16*)p;         // K = k+bbk .. +7
      bfr[j].h[1] = *(const v8u16*)(p + 16);  // K = k+16+bbk.. +7
    }
#pragma unroll
    for (int i = 0; i < 4; ++i)
#pragma unroll
      for (int j = 0; j < 4; ++j)
        acc[i][j] = __builtin_amdgcn_wmma_f32_16x16x32_bf16(
            false, afr[i].bf, false, bfr[j].bf, (short)0, acc[i][j],
            false, false);
  }

  // C/D layout: lane L holds column N=L&15, rows M = r + 8*(L>>4), r=0..7
  unsigned short* Od = Out + (size_t)d * M * N;
  const int mr = (lane >> 4) * 8;
#pragma unroll
  for (int i = 0; i < 4; ++i)
#pragma unroll
    for (int j = 0; j < 4; ++j)
#pragma unroll
      for (int r = 0; r < 8; ++r) {
        int m = m0 + 16 * i + mr + r;
        Od[(size_t)m * N + n0 + 16 * j + lm] = f2bf(acc[i][j][r]);
      }
}

// ---------------------------------------------------------------------------
// 4) BN statistics: deterministic fixed-order partial sums (no atomics).
//    grid = (CH_ chunks, D_), block 256; each chunk reduces B_/CH_ rows.
// ---------------------------------------------------------------------------
__global__ void stats_kernel(const unsigned short* __restrict__ H, int N,
                             int rowsPerChunk, float* __restrict__ psum,
                             float* __restrict__ psq, int Bn) {
  int d = blockIdx.y;
  int chunk = blockIdx.x;
  const unsigned short* Hd =
      H + ((size_t)d * Bn + (size_t)chunk * rowsPerChunk) * N;
  int total = gridDim.y * N;  // D*N
  for (int n = threadIdx.x; n < N; n += blockDim.x) {
    float s = 0.f, q = 0.f;
    for (int r = 0; r < rowsPerChunk; ++r) {
      float x = bf2f(Hd[(size_t)r * N + n]);
      s += x;
      q += x * x;
    }
    psum[(size_t)chunk * total + d * N + n] = s;
    psq[(size_t)chunk * total + d * N + n] = q;
  }
}

// ---------------------------------------------------------------------------
// 5) BN coefficient prep: a = gamma*rsqrt(var+eps), c = beta - mean*a.
//    (layer biases b1/b2 cancel inside BN and are dropped entirely)
// ---------------------------------------------------------------------------
__global__ void bn_prep_kernel(const float* __restrict__ psum,
                               const float* __restrict__ psq,
                               const float* __restrict__ gamma,
                               const float* __restrict__ beta,
                               float* __restrict__ a, float* __restrict__ c,
                               int total, int chunks, float invB) {
  int i = blockIdx.x * blockDim.x + threadIdx.x;
  if (i >= total) return;
  float s = 0.f, q = 0.f;
  for (int ch = 0; ch < chunks; ++ch) {
    s += psum[(size_t)ch * total + i];
    q += psq[(size_t)ch * total + i];
  }
  float m = s * invB;
  float v = fmaxf(q * invB - m * m, 0.f);
  float sc = gamma[i] * rsqrtf(v + 1e-5f);
  a[i] = sc;
  c[i] = beta[i] - m * sc;
}

// ---------------------------------------------------------------------------
// 6) BN+ReLU apply in place (h1 only; h2's BN folds into the final kernel).
// ---------------------------------------------------------------------------
__global__ void bn_relu_kernel(unsigned short* __restrict__ H,
                               const float* __restrict__ a,
                               const float* __restrict__ c, int N,
                               int rowsPerD, size_t nvec) {
  size_t idx = (size_t)blockIdx.x * blockDim.x + threadIdx.x;
  if (idx >= nvec) return;
  size_t base = idx * 8;
  size_t row = base / N;
  int n = (int)(base - row * N);
  int d = (int)(row / rowsPerD);
  const float* ad = a + (size_t)d * N + n;
  const float* cd = c + (size_t)d * N + n;
  v8u16 x = *(v8u16*)(H + base);
  v4f a0 = *(const v4f*)ad, a1v = *(const v4f*)(ad + 4);
  v4f c0 = *(const v4f*)cd, c1v = *(const v4f*)(cd + 4);
  v8u16 o;
#pragma unroll
  for (int e = 0; e < 8; ++e) {
    float aa = (e < 4) ? a0[e] : a1v[e - 4];
    float cc = (e < 4) ? c0[e] : c1v[e - 4];
    float y = fmaxf(fmaf(bf2f(x[e]), aa, cc), 0.f);
    o[e] = f2bf(y);
  }
  *(v8u16*)(H + base) = o;
}

// ---------------------------------------------------------------------------
// 7) Final: per sample select its domain, apply BN2+ReLU on the fly, dot with
//    W3, add b3, sigmoid. One wave per sample, lane-strided K, shuffle reduce.
// ---------------------------------------------------------------------------
__global__ __launch_bounds__(256)
void final_kernel(const unsigned short* __restrict__ H2,
                  const float* __restrict__ a2, const float* __restrict__ c2,
                  const float* __restrict__ W3, const float* __restrict__ b3,
                  const int* __restrict__ domain_id, float* __restrict__ out,
                  int Bn, int N) {
  int lane = threadIdx.x & 31;
  int wave = threadIdx.x >> 5;
  int b = blockIdx.x * 8 + wave;
  if (b >= Bn) return;
  int d = domain_id[b];
  const unsigned short* h = H2 + ((size_t)d * Bn + b) * N;
  const float* ad = a2 + (size_t)d * N;
  const float* cd = c2 + (size_t)d * N;
  const float* wd = W3 + (size_t)d * N;
  float acc = 0.f;
  for (int k = lane; k < N; k += 32) {
    float x = bf2f(h[k]);
    float v = fmaxf(fmaf(x, ad[k], cd[k]), 0.f);
    acc = fmaf(v, wd[k], acc);
  }
#pragma unroll
  for (int off = 16; off > 0; off >>= 1) acc += __shfl_down(acc, off, 32);
  if (lane == 0) out[b] = 1.f / (1.f + expf(-(acc + b3[d])));
}

// ---------------------------------------------------------------------------
// Launch: gather/cast -> GEMM1 -> stats1 -> prep1 -> bnrelu1 -> GEMM2 ->
//         stats2 -> prep2 -> final.  Workspace ~228 MB.
// ---------------------------------------------------------------------------
extern "C" void kernel_launch(void* const* d_in, const int* in_sizes, int n_in,
                              void* d_out, int out_size, void* d_ws,
                              size_t ws_size, hipStream_t stream) {
  (void)in_sizes; (void)n_in; (void)out_size; (void)ws_size;
  const int*   feat_ids  = (const int*)d_in[0];
  const int*   domain_id = (const int*)d_in[1];
  const float* emb_tab   = (const float*)d_in[2];
  const float* W1  = (const float*)d_in[3];
  // d_in[4] = b1 : cancels under BatchNorm, unused
  const float* g1  = (const float*)d_in[5];
  const float* be1 = (const float*)d_in[6];
  const float* W2  = (const float*)d_in[7];
  // d_in[8] = b2 : cancels under BatchNorm, unused
  const float* g2  = (const float*)d_in[9];
  const float* be2 = (const float*)d_in[10];
  const float* W3  = (const float*)d_in[11];
  const float* b3  = (const float*)d_in[12];
  float* out = (float*)d_out;

  // workspace carve-up (all 16B-aligned)
  unsigned short* embB = (unsigned short*)d_ws;               // B*IN bf16
  unsigned short* W1B  = embB + (size_t)B_ * IN_;             // D*H1*IN bf16
  unsigned short* W2B  = W1B + (size_t)D_ * H1_ * IN_;        // D*H2*H1 bf16
  unsigned short* H1P  = W2B + (size_t)D_ * H2_ * H1_;        // D*B*H1 bf16
  unsigned short* H2P  = H1P + (size_t)D_ * B_ * H1_;         // D*B*H2 bf16
  float* psum1 = (float*)(H2P + (size_t)D_ * B_ * H2_);       // CH*D*H1
  float* psq1  = psum1 + (size_t)CH_ * D_ * H1_;
  float* psum2 = psq1  + (size_t)CH_ * D_ * H1_;              // CH*D*H2
  float* psq2  = psum2 + (size_t)CH_ * D_ * H2_;
  float* a1 = psq2 + (size_t)CH_ * D_ * H2_;
  float* c1 = a1 + (size_t)D_ * H1_;
  float* a2 = c1 + (size_t)D_ * H1_;
  float* c2 = a2 + (size_t)D_ * H2_;

  // weight casts + embedding gather
  {
    size_t n8 = (size_t)D_ * H1_ * IN_ / 8;
    cast_bf16_kernel<<<dim3((unsigned)((n8 + 255) / 256)), 256, 0, stream>>>(
        W1, W1B, n8);
    n8 = (size_t)D_ * H2_ * H1_ / 8;
    cast_bf16_kernel<<<dim3((unsigned)((n8 + 255) / 256)), 256, 0, stream>>>(
        W2, W2B, n8);
  }
  gather_cast_kernel<<<dim3(B_ * F_ / 256), 256, 0, stream>>>(feat_ids,
                                                              emb_tab, embB);

  // layer 1: h1_pre = emb @ W1[d]^T    (A shared across domains: stride 0)
  gemm_bf16<<<dim3(B_ / 64, H1_ / 512, D_), 256, 0, stream>>>(
      embB, 0, W1B, (size_t)H1_ * IN_, H1P, B_, H1_, IN_);
  stats_kernel<<<dim3(CH_, D_), 256, 0, stream>>>(H1P, H1_, B_ / CH_, psum1,
                                                  psq1, B_);
  bn_prep_kernel<<<dim3((D_ * H1_ + 255) / 256), 256, 0, stream>>>(
      psum1, psq1, g1, be1, a1, c1, D_ * H1_, CH_, 1.f / B_);
  {
    size_t nvec = (size_t)D_ * B_ * H1_ / 8;
    bn_relu_kernel<<<dim3((unsigned)((nvec + 255) / 256)), 256, 0, stream>>>(
        H1P, a1, c1, H1_, B_, nvec);
  }

  // layer 2: h2_pre = h1 @ W2[d]^T     (A per-domain: stride B*H1)
  gemm_bf16<<<dim3(B_ / 64, H2_ / 512, D_), 256, 0, stream>>>(
      H1P, (size_t)B_ * H1_, W2B, (size_t)H2_ * H1_, H2P, B_, H2_, H1_);
  stats_kernel<<<dim3(CH_, D_), 256, 0, stream>>>(H2P, H2_, B_ / CH_, psum2,
                                                  psq2, B_);
  bn_prep_kernel<<<dim3((D_ * H2_ + 255) / 256), 256, 0, stream>>>(
      psum2, psq2, g2, be2, a2, c2, D_ * H2_, CH_, 1.f / B_);

  // final: BN2+ReLU (selected domain only) + dot W3 + b3 + sigmoid
  final_kernel<<<dim3(B_ / 8), 256, 0, stream>>>(H2P, a2, c2, W3, b3,
                                                 domain_id, out, B_, H2_);
}
